// LocationAwareAttention_83305185673310
// MI455X (gfx1250) — matrix operand order repaired and checked by reading, not compiled, not run on hardware
//
#include <hip/hip_runtime.h>
#include <hip/hip_bf16.h>
#include <math.h>

#define B_   64
#define T_   2048
#define ENCD 512
#define DECD 1024
#define ATT  256
#define NF   32
#define KS   31

typedef __attribute__((ext_vector_type(16))) __bf16 v16bf;
typedef __attribute__((ext_vector_type(8)))  float  v8f;

__device__ __forceinline__ v8f wmma_bf16(v16bf a, v16bf b, v8f c) {
  // D = A(16x32 bf16) * B(32x16 bf16) + C(16x16 f32)
  return __builtin_amdgcn_wmma_f32_16x16x32_bf16(false, a, false, b, (short)0, c,
                                                 false, false);
}

__device__ __forceinline__ void split_bf16(const float* x, v16bf& hi, v16bf& lo) {
#pragma unroll
  for (int i = 0; i < 16; ++i) {
    __bf16 h = (__bf16)x[i];
    hi[i] = h;
    lo[i] = (__bf16)(x[i] - (float)h);
  }
}

// ---------------------------------------------------------------------------
// Prep 1: dec_p[b][a] = sum_d decoder_hidden[b][d] * W_dec[a][d]
// ---------------------------------------------------------------------------
__global__ void __launch_bounds__(256)
k_prep_dec(const float* __restrict__ dh, const float* __restrict__ Wdec,
           float* __restrict__ decp) {
  int b = blockIdx.x, a = threadIdx.x;
  const float4* x = (const float4*)(dh + (size_t)b * DECD);
  const float4* w = (const float4*)(Wdec + (size_t)a * DECD);
  float acc = 0.0f;
  for (int i = 0; i < DECD / 4; ++i) {
    float4 xv = x[i], wv = w[i];
    acc += xv.x * wv.x + xv.y * wv.y + xv.z * wv.z + xv.w * wv.w;
  }
  decp[b * ATT + a] = acc;
}

// ---------------------------------------------------------------------------
// Prep 2: swizzle W_enc (ATT x ENC, fp32) into WMMA B-fragment order, bf16 hi/lo.
// frag linear index = (((nt*16 + kk)*32) + lane)*16 + e
// lane = kh*16 + nl holds B[k = kk*32 + kh*16 + e][n = nt*16 + nl] = W_enc[n][k]
// ---------------------------------------------------------------------------
__global__ void __launch_bounds__(256)
k_prep_wenc(const float* __restrict__ Wenc,
            __bf16* __restrict__ hi, __bf16* __restrict__ lo) {
  int idx = blockIdx.x * 256 + threadIdx.x;   // 16*16*32*16 = 131072
  int e    = idx & 15;
  int lane = (idx >> 4) & 31;
  int kk   = (idx >> 9) & 15;
  int nt   = idx >> 13;
  int kh = lane >> 4, nl = lane & 15;
  float v = Wenc[(size_t)(nt * 16 + nl) * ENCD + kk * 32 + kh * 16 + e];
  __bf16 h = (__bf16)v;
  hi[idx] = h;
  lo[idx] = (__bf16)(v - (float)h);
}

// ---------------------------------------------------------------------------
// Prep 3: Wla[n][k] = sum_f W_loc[n][f] * W_conv[f][k]  (k < 31, else 0)
// stored as the extra K-step B fragment (hi/lo), index = ((nt*32)+lane)*16 + e
// ---------------------------------------------------------------------------
__global__ void __launch_bounds__(256)
k_prep_wla(const float* __restrict__ Wloc, const float* __restrict__ Wconv,
           __bf16* __restrict__ hi, __bf16* __restrict__ lo) {
  int idx = blockIdx.x * 256 + threadIdx.x;   // 16*32*16 = 8192
  int e    = idx & 15;
  int lane = (idx >> 4) & 31;
  int nt   = idx >> 9;
  int kh = lane >> 4, nl = lane & 15;
  int n = nt * 16 + nl;
  int k = kh * 16 + e;
  float v = 0.0f;
  if (k < KS) {
    for (int f = 0; f < NF; ++f) v += Wloc[n * NF + f] * Wconv[f * KS + k];
  }
  __bf16 h = (__bf16)v;
  hi[idx] = h;
  lo[idx] = (__bf16)(v - (float)h);
}

// ---------------------------------------------------------------------------
// Main energy kernel.
// Block = 8 waves; each wave owns 16 rows (t); N=256 swept in two chunks of
// 8 n-tiles; K = 16 steps of 32 (encoder) + 1 im2col step (location conv).
// B fragments (hi/lo) for the current K-step are staged in LDS once per block
// via async global->LDS copies (double-buffered, ASYNCcnt + barrier), so the
// 8 waves share one L2 read instead of eight.
// ---------------------------------------------------------------------------
#define NSTEP 34   // 2 ntc passes * 17 K-steps
// per-step B slab: [hilo][nt(8)][lane(32)][e(16)] bf16 = 8192 elems = 16 KB
__device__ __forceinline__ void stage_step(int s, __bf16* dstBase,
                                           const __bf16* fBhi, const __bf16* fBlo,
                                           const __bf16* fLhi, const __bf16* fLlo) {
  const int ntc = s / 17, kk = s % 17;
  const int tid = threadIdx.x;
#pragma unroll
  for (int i = 0; i < 4; ++i) {
    int c    = tid + i * 256;        // 1024 chunks of 16B
    int hl   = c >> 9;
    int c2   = c & 511;
    int nt   = c2 >> 6;
    int rest = c2 & 63;
    int lane = rest >> 1;
    int e8   = (rest & 1) * 8;
    int ntg  = ntc * 8 + nt;
    const __bf16* src;
    if (kk < 16) {
      const __bf16* base = hl ? fBlo : fBhi;
      src = base + ((size_t)(ntg * 16 + kk) * 32 + lane) * 16 + e8;
    } else {
      const __bf16* base = hl ? fLlo : fLhi;
      src = base + ((size_t)ntg * 32 + lane) * 16 + e8;
    }
    __bf16* dst = dstBase + ((hl * 8 + nt) * 32 + lane) * 16 + e8;
    // flat LDS aperture: addr[31:0] is the wave-relative LDS byte offset
    unsigned int       ldsOff = (unsigned int)(uintptr_t)(void*)dst;
    unsigned long long gaddr  = (unsigned long long)(uintptr_t)(const void*)src;
    asm volatile("global_load_async_to_lds_b128 %0, %1, off"
                 :: "v"(ldsOff), "v"(gaddr) : "memory");
  }
}

__global__ void __launch_bounds__(256)
k_energy(const float* __restrict__ enc, const float* __restrict__ prev,
         const __bf16* __restrict__ fragBhi, const __bf16* __restrict__ fragBlo,
         const __bf16* __restrict__ fragLhi, const __bf16* __restrict__ fragLlo,
         const float* __restrict__ decp, const float* __restrict__ Wen,
         float* __restrict__ energy) {
  __shared__ __bf16 shB[2][2 * 8 * 32 * 16];   // double buffer, 2 x 16 KB

  const int b    = blockIdx.x >> 4;
  const int t0   = (blockIdx.x & 15) * 128;
  const int w    = threadIdx.x >> 5;
  const int lane = threadIdx.x & 31;
  const int kh   = lane >> 4;
  const int nl   = lane & 15;
  const int mrow = lane & 15;            // A-fragment row handled by this lane
  const int tA   = t0 + w * 16 + mrow;   // its global t

  const float* Arow = enc + ((size_t)b * T_ + tA) * ENCD;
  const float* prow = prev + (size_t)b * T_;

  // im2col A fragment for the conv K-step: element e -> tap j, value prev[tA+j-15]
  float jv[16];
#pragma unroll
  for (int e = 0; e < 16; ++e) {
    int j = kh * 16 + e;
    int g = tA + j - 15;
    jv[e] = (j < KS && g >= 0 && g < T_) ? prow[g] : 0.0f;
  }
  v16bf ph, pl;
  split_bf16(jv, ph, pl);

  float eacc[8];
#pragma unroll
  for (int r = 0; r < 8; ++r) eacc[r] = 0.0f;

  // pipeline prologue: stage step 0 into buffer 0
  stage_step(0, &shB[0][0], fragBhi, fragBlo, fragLhi, fragLlo);
  __builtin_amdgcn_s_wait_asynccnt(0);
  __syncthreads();

  int s = 0;
  for (int ntc = 0; ntc < 2; ++ntc) {
    v8f acc[8] = {};

    for (int kk = 0; kk < 17; ++kk, ++s) {
      const int cur = s & 1;
      if (s + 1 < NSTEP)
        stage_step(s + 1, &shB[cur ^ 1][0], fragBhi, fragBlo, fragLhi, fragLlo);

      // A fragment: elems 0..15 -> K = kh*16+0..15
      v16bf ah, al;
      if (kk < 16) {
        float av[16];
        const float* ap = Arow + kk * 32 + kh * 16;
        *(float4*)(av + 0)  = *(const float4*)(ap);
        *(float4*)(av + 4)  = *(const float4*)(ap + 4);
        *(float4*)(av + 8)  = *(const float4*)(ap + 8);
        *(float4*)(av + 12) = *(const float4*)(ap + 12);
        split_bf16(av, ah, al);
      } else {
        ah = ph;
        al = pl;
      }

      const __bf16* shPtr = &shB[cur][0];
#pragma unroll
      for (int nt = 0; nt < 8; ++nt) {
        v16bf bh = *(const v16bf*)(shPtr + ((size_t)nt * 32 + lane) * 16);
        v16bf bl = *(const v16bf*)(shPtr + ((size_t)(8 + nt) * 32 + lane) * 16);
        acc[nt] = wmma_bf16(ah, bh, acc[nt]);
        acc[nt] = wmma_bf16(al, bh, acc[nt]);
        acc[nt] = wmma_bf16(ah, bl, acc[nt]);
      }

      __builtin_amdgcn_s_wait_asynccnt(0);  // next slab landed in LDS
      __syncthreads();                      // all waves done reading cur slab
    }

    // epilogue: C layout -> VGPR r, lanes 0-15: M=r, lanes 16-31: M=8+r; N=nl
#pragma unroll
    for (int nt = 0; nt < 8; ++nt) {
      int n = (ntc * 8 + nt) * 16 + nl;
      float wE = Wen[n];
      float dp = decp[b * ATT + n];
#pragma unroll
      for (int r = 0; r < 8; ++r)
        eacc[r] += wE * tanhf(acc[nt][r] + dp);
    }
  }

  // sum over the 16 lanes of each half (same row set, different n mod 16)
#pragma unroll
  for (int r = 0; r < 8; ++r) {
    float v = eacc[r];
    v += __shfl_xor(v, 1, 32);
    v += __shfl_xor(v, 2, 32);
    v += __shfl_xor(v, 4, 32);
    v += __shfl_xor(v, 8, 32);
    eacc[r] = v;
  }
  if (nl == 0) {
#pragma unroll
    for (int r = 0; r < 8; ++r) {
      int t = t0 + w * 16 + kh * 8 + r;
      energy[(size_t)b * T_ + t] = eacc[r];
    }
  }
}

// ---------------------------------------------------------------------------
// Softmax over T per batch row (mask applied here)
// ---------------------------------------------------------------------------
__global__ void __launch_bounds__(256)
k_softmax(const float* __restrict__ energy, const int* __restrict__ mask,
          float* __restrict__ attn) {
  int b = blockIdx.x, tid = threadIdx.x;
  __shared__ float red[256];
  float ev[8];
  float lmax = -3.0e38f;
#pragma unroll
  for (int i = 0; i < 8; ++i) {
    int t = tid + i * 256;
    float e = energy[(size_t)b * T_ + t];
    if (mask[(size_t)b * T_ + t] == 0) e = -1e9f;
    ev[i] = e;
    lmax = fmaxf(lmax, e);
  }
  red[tid] = lmax;
  __syncthreads();
  for (int s = 128; s > 0; s >>= 1) {
    if (tid < s) red[tid] = fmaxf(red[tid], red[tid + s]);
    __syncthreads();
  }
  float gmax = red[0];
  __syncthreads();
  float lsum = 0.0f;
#pragma unroll
  for (int i = 0; i < 8; ++i) {
    ev[i] = expf(ev[i] - gmax);
    lsum += ev[i];
  }
  red[tid] = lsum;
  __syncthreads();
  for (int s = 128; s > 0; s >>= 1) {
    if (tid < s) red[tid] += red[tid + s];
    __syncthreads();
  }
  float inv = 1.0f / red[0];
#pragma unroll
  for (int i = 0; i < 8; ++i)
    attn[(size_t)b * T_ + tid + i * 256] = ev[i] * inv;
}

// ---------------------------------------------------------------------------
// context partials: grid (B, 4); each block reduces a 512-t chunk
// ---------------------------------------------------------------------------
__global__ void __launch_bounds__(256)
k_ctx_part(const float* __restrict__ enc, const float* __restrict__ attn,
           float* __restrict__ part) {
  int b = blockIdx.x, c = blockIdx.y, tid = threadIdx.x;
  const float* encb = enc + ((size_t)b * T_ + (size_t)c * 512) * ENCD;
  const float* at   = attn + (size_t)b * T_ + (size_t)c * 512;
  float acc0 = 0.0f, acc1 = 0.0f;
  for (int t = 0; t < 512; ++t) {
    float a = at[t];
    const float* row = encb + (size_t)t * ENCD;
    acc0 += a * row[tid];
    acc1 += a * row[tid + 256];
  }
  part[((size_t)(b * 4 + c)) * ENCD + tid]       = acc0;
  part[((size_t)(b * 4 + c)) * ENCD + tid + 256] = acc1;
}

__global__ void __launch_bounds__(256)
k_ctx_reduce(const float* __restrict__ part, float* __restrict__ ctx) {
  int idx = blockIdx.x * 256 + threadIdx.x;  // 0 .. B*ENC-1
  int b = idx >> 9, e = idx & 511;
  float s = 0.0f;
  for (int c = 0; c < 4; ++c) s += part[((size_t)(b * 4 + c)) * ENCD + e];
  ctx[idx] = s;
}

// ---------------------------------------------------------------------------
extern "C" void kernel_launch(void* const* d_in, const int* in_sizes, int n_in,
                              void* d_out, int out_size, void* d_ws, size_t ws_size,
                              hipStream_t stream) {
  (void)in_sizes; (void)n_in; (void)out_size; (void)ws_size;
  const float* dh    = (const float*)d_in[0];  // (B, DEC)
  const float* enc   = (const float*)d_in[1];  // (B, T, ENC)
  const float* prev  = (const float*)d_in[2];  // (B, T)
  const int*   mask  = (const int*)d_in[3];    // (B, T)
  const float* Wenc  = (const float*)d_in[4];  // (ATT, ENC)
  const float* Wdec  = (const float*)d_in[5];  // (ATT, DEC)
  const float* Wconv = (const float*)d_in[6];  // (NF, 1, KS)
  const float* Wloc  = (const float*)d_in[7];  // (ATT, NF)
  const float* Wen   = (const float*)d_in[8];  // (1, ATT)

  char* ws = (char*)d_ws;
  __bf16* fragBhi = (__bf16*)(ws + 0);        // 262144 B
  __bf16* fragBlo = (__bf16*)(ws + 262144);   // 262144 B
  __bf16* fragLhi = (__bf16*)(ws + 524288);   //  16384 B
  __bf16* fragLlo = (__bf16*)(ws + 540672);   //  16384 B
  float*  decp    = (float*)(ws + 557056);    //  65536 B
  float*  energy  = (float*)(ws + 622592);    // 524288 B
  float*  part    = (float*)(ws + 1146880);   // 524288 B

  float* ctxOut  = (float*)d_out;             // (B, ENC)
  float* attnOut = (float*)d_out + B_ * ENCD; // (B, T)

  k_prep_dec <<<B_, 256, 0, stream>>>(dh, Wdec, decp);
  k_prep_wenc<<<512, 256, 0, stream>>>(Wenc, fragBhi, fragBlo);
  k_prep_wla <<<32, 256, 0, stream>>>(Wloc, Wconv, fragLhi, fragLlo);

  k_energy<<<B_ * (T_ / 128), 256, 0, stream>>>(enc, prev, fragBhi, fragBlo,
                                                fragLhi, fragLlo, decp, Wen,
                                                energy);

  k_softmax<<<B_, 256, 0, stream>>>(energy, mask, attnOut);

  k_ctx_part<<<dim3(B_, 4), 256, 0, stream>>>(enc, attnOut, part);
  k_ctx_reduce<<<(B_ * ENCD) / 256, 256, 0, stream>>>(part, ctxOut);
}
